// MySatNerf_670014898724
// MI455X (gfx1250) — matrix operand order repaired and checked by reading, not compiled, not run on hardware
//
#include <hip/hip_runtime.h>
#include <hip/hip_bf16.h>

typedef _Float16 f16;
typedef __attribute__((ext_vector_type(16))) _Float16 v16h;
typedef __attribute__((ext_vector_type(8)))  _Float16 v8h;
typedef __attribute__((ext_vector_type(8)))  float    v8f;

#define WAVES 8
#define ROWS  32            // points per wave (two 16-row M tiles per WMMA B-frag)
#define SA 264              // halves per row, buffer A (256 cols + pad, 16B-aligned rows)
#define SB 328              // halves per row, buffer B (320 cols + pad)
#define WAVE_HALVES (ROWS * SA + ROWS * SB)
#define LDS_HALVES (WAVES * WAVE_HALVES)
#define LDS_BYTES  (LDS_HALVES * 2)   // 303,104 B <= 320KB WGP limit

enum { EPI_SIN = 0, EPI_SINSIN = 1, EPI_L7 = 2, EPI_SOFTPLUS = 3,
       EPI_RAW3 = 4, EPI_SIG3 = 5, EPI_SIG1 = 6 };

struct NerfParams {
  const float* x;
  const float* w;
  const f16*   Wsp[8];
  const float* bsp[8];
  const f16 *Wmid, *Walb, *Wunc, *Wamb0, *Wamb1, *Wsh0, *Wsh1, *Wsh2;
  const float *bmid, *balb, *bunc, *bamb0, *bamb1, *bsh0, *bsh1, *bsh2;
  float* out;
  int n;
};

// ---------------------------------------------------------------- prep kernel
// Repack an f32 (K,N) row-major weight matrix into f16 (Kp,Np) row-major with
// zero padding. B-frag layout for v_wmma_f32_16x16x32_f16: lane l holds column
// N=nt*16+l%16, halves 0..15 = K rows kt*32+8h+{0..7,16..23} -> Np-strided in this layout.
__global__ void pack_w(const float* __restrict__ src, f16* __restrict__ dst,
                       int K, int N, int Kp, int Np) {
  int idx = blockIdx.x * blockDim.x + threadIdx.x;
  int total = Kp * Np;
  if (idx >= total) return;
  int r = idx / Np, c = idx - r * Np;
  float v = (r < K && c < N) ? src[r * N + c] : 0.0f;
  dst[idx] = (f16)v;
}

// ---------------------------------------------------------------- device utils
// One row per lane (32 rows per wave); column index is compile-time so the
// 2^l scale folds to a literal; stores as 16B v8h chunks.
__device__ __forceinline__ void fill_posenc32(f16* buf, int stride, int colBase,
                                              const float* src, int pb, bool extraSin) {
  int lane = threadIdx.x & 31;
  const float* xv = src + (size_t)(pb + lane) * 3;
  float v0 = xv[0], v1 = xv[1], v2 = xv[2];
  f16* dst = buf + lane * stride + colBase;
#pragma unroll
  for (int ch = 0; ch < 8; ++ch) {
    v8h pk;
#pragma unroll
    for (int j = 0; j < 8; ++j) {
      int c = ch * 8 + j;
      float val = 0.0f;
      if (c < 60) {
        int l = c / 6;
        int r6 = c - l * 6;
        int d = (r6 >= 3) ? (r6 - 3) : r6;
        float vv = (d == 0) ? v0 : ((d == 1) ? v1 : v2);
        float ang = vv * (float)(1 << l);          // 2^l, l in [0,9]
        val = (r6 < 3) ? __sinf(ang) : __cosf(ang);
        if (extraSin) val = __sinf(val);
      }
      pk[j] = (f16)val;
    }
    *(v8h*)(dst + ch * 8) = pk;
  }
}

template <int EPI>
__device__ __forceinline__ void epi_store(v8f acc, int localBase, int globalBase,
                                          int col, int col_lo,
                                          f16* outBuf, int outS, float* out) {
  if (EPI == EPI_SIN || EPI == EPI_SINSIN || EPI == EPI_L7) {
#pragma unroll
    for (int r = 0; r < 8; ++r) {
      float v = acc[r];
      float sv;
      if (EPI == EPI_L7) {
        if (col == 255) { out[globalBase + r] = v; sv = 0.0f; }  // density; zero pad col
        else sv = __sinf(v);
      } else if (EPI == EPI_SINSIN) {
        sv = __sinf(__sinf(v));
      } else {
        sv = __sinf(v);
      }
      outBuf[(localBase + r) * outS + col] = (f16)sv;
    }
  } else {
#pragma unroll
    for (int r = 0; r < 8; ++r) {
      float v = acc[r];
      int row = globalBase + r;
      if (EPI == EPI_SOFTPLUS) {
        if (col_lo == 0) out[row] = (v > 20.0f) ? v : __logf(1.0f + __expf(v));
      }
      if (EPI == EPI_RAW3) {
        if (col_lo < 3) out[(size_t)row * 3 + col_lo] = v;
      }
      if (EPI == EPI_SIG3) {
        if (col_lo < 3) out[(size_t)row * 3 + col_lo] = 1.0f / (1.0f + __expf(-v));
      }
      if (EPI == EPI_SIG1) {
        if (col_lo == 0) out[row] = 1.0f / (1.0f + __expf(-v));
      }
    }
  }
}

// One 32xK @ KxN layer for one wave's 32-point tile (two 16-row M tiles so
// every B-fragment load feeds two v_wmma).
template <int KT, int EPI>
__device__ __forceinline__ void do_layer(const f16* in, int inS, int colBase,
                                         f16* outBuf, int outS,
                                         const f16* __restrict__ W, int Np, int NT,
                                         const float* __restrict__ bias, int nbias,
                                         float* out, int pb) {
  int lane   = threadIdx.x & 31;
  int col_lo = lane & 15;
  int hi8    = (lane >> 4) << 3;   // 0 for lanes 0-15, 8 for lanes 16-31

  // A-fragments: lane holds row (lane&15); halves = K {b..b+7, b+16..b+23},
  // b = kt*32 + 8*(lane>=16)  ->  two aligned 16B LDS chunks per fragment.
  v16h af0[KT], af1[KT];
  {
    const f16* rp0 = in + colBase + col_lo * inS + hi8;
    const f16* rp1 = rp0 + 16 * inS;
#pragma unroll
    for (int kt = 0; kt < KT; ++kt) {
      v8h lo0 = *(const v8h*)(rp0 + kt * 32);
      v8h hi0 = *(const v8h*)(rp0 + kt * 32 + 16);
      af0[kt] = __builtin_shufflevector(lo0, hi0, 0, 1, 2, 3, 4, 5, 6, 7,
                                        8, 9, 10, 11, 12, 13, 14, 15);
      v8h lo1 = *(const v8h*)(rp1 + kt * 32);
      v8h hi1 = *(const v8h*)(rp1 + kt * 32 + 16);
      af1[kt] = __builtin_shufflevector(lo1, hi1, 0, 1, 2, 3, 4, 5, 6, 7,
                                        8, 9, 10, 11, 12, 13, 14, 15);
    }
  }

  const f16* wlane = W + col_lo;   // B-frag: lane = N column

  for (int nt = 0; nt < NT; ++nt) {
    int col = nt * 16 + col_lo;
    float bval = (col < nbias) ? bias[col] : 0.0f;
    v8f acc0 = {bval, bval, bval, bval, bval, bval, bval, bval};
    v8f acc1 = acc0;
    const f16* wp = wlane + nt * 16;
#pragma unroll
    for (int kt = 0; kt < KT; ++kt) {
      v16h b;
#pragma unroll
      for (int s = 0; s < 16; ++s)
        b[s] = wp[(size_t)(kt * 32 + hi8 + (s & 7) + ((s >> 3) << 4)) * Np];
      acc0 = __builtin_amdgcn_wmma_f32_16x16x32_f16(
          false, af0[kt], false, b, (short)0, acc0, false, false);
      acc1 = __builtin_amdgcn_wmma_f32_16x16x32_f16(
          false, af1[kt], false, b, (short)0, acc1, false, false);
    }
    epi_store<EPI>(acc0, hi8,      pb + hi8,      col, col_lo, outBuf, outS, out);
    epi_store<EPI>(acc1, 16 + hi8, pb + 16 + hi8, col, col_lo, outBuf, outS, out);
  }
}

// ---------------------------------------------------------------- main kernel
__global__ void __launch_bounds__(WAVES * 32) nerf_main(NerfParams P) {
  extern __shared__ __align__(16) f16 smem[];
  int wave = threadIdx.x >> 5;
  f16* bufA = smem + (size_t)wave * WAVE_HALVES;
  f16* bufB = bufA + ROWS * SA;
  int pb = (blockIdx.x * WAVES + wave) * ROWS;   // point base for this wave

  float* o_dens = P.out;
  float* o_unc  = P.out + (size_t)P.n;
  float* o_alb  = P.out + (size_t)2 * P.n;
  float* o_shad = P.out + (size_t)5 * P.n;
  float* o_amb  = P.out + (size_t)6 * P.n;

  // posenc(x) -> bufA cols 0..63 (raw enc); sin(posenc(w)) -> bufB cols 256..319
  fill_posenc32(bufA, SA, 0,   P.x, pb, false);
  fill_posenc32(bufB, SB, 256, P.w, pb, true);

  // spatial MLP, ping-pong A<->B (cols 0..255 only, so sw in B survives)
  do_layer<2, EPI_SIN>(bufA, SA, 0, bufB, SB, P.Wsp[0], 256, 16, P.bsp[0], 256, nullptr, pb);
  do_layer<8, EPI_SIN>(bufB, SB, 0, bufA, SA, P.Wsp[1], 256, 16, P.bsp[1], 256, nullptr, pb);
  do_layer<8, EPI_SIN>(bufA, SA, 0, bufB, SB, P.Wsp[2], 256, 16, P.bsp[2], 256, nullptr, pb);
  do_layer<8, EPI_SIN>(bufB, SB, 0, bufA, SA, P.Wsp[3], 256, 16, P.bsp[3], 256, nullptr, pb);
  do_layer<8, EPI_SIN>(bufA, SA, 0, bufB, SB, P.Wsp[4], 256, 16, P.bsp[4], 256, nullptr, pb);
  do_layer<8, EPI_SIN>(bufB, SB, 0, bufA, SA, P.Wsp[5], 256, 16, P.bsp[5], 256, nullptr, pb);
  do_layer<8, EPI_SIN>(bufA, SA, 0, bufB, SB, P.Wsp[6], 256, 16, P.bsp[6], 256, nullptr, pb);
  // L7 (linear): density = col 255; store sin(m) with col255 zeroed -> A
  do_layer<8, EPI_L7>(bufB, SB, 0, bufA, SA, P.Wsp[7], 256, 16, P.bsp[7], 256, o_dens, pb);
  // middle (W padded 255->256 with zero row): store sin(sin(.)) = sin(m2) -> B cols 0..255
  do_layer<8, EPI_SINSIN>(bufA, SA, 0, bufB, SB, P.Wmid, 256, 16, P.bmid, 256, nullptr, pb);
  // uncertainty head: sw (B cols 256..319, K padded 60->64) -> softplus
  do_layer<2, EPI_SOFTPLUS>(bufB, SB, 256, nullptr, 0, P.Wunc, 16, 1, P.bunc, 1, o_unc, pb);
  // ambient[0]: sw -> sin -> A cols 0..127
  do_layer<2, EPI_SIN>(bufB, SB, 256, bufA, SA, P.Wamb0, 128, 8, P.bamb0, 128, nullptr, pb);
  // ambient[1]: A cols 0..127 -> raw 3 outputs
  do_layer<4, EPI_RAW3>(bufA, SA, 0, nullptr, 0, P.Wamb1, 16, 1, P.bamb1, 3, o_amb, pb);
  // albedo: sin(m2) (B cols 0..255) -> sigmoid 3 outputs
  do_layer<8, EPI_SIG3>(bufB, SB, 0, nullptr, 0, P.Walb, 16, 1, P.balb, 3, o_alb, pb);
  // shading[0]: h = [sin(m2)|sw|0] (B, 320 cols, W padded 316->320) -> sin -> A cols 0..127
  do_layer<10, EPI_SIN>(bufB, SB, 0, bufA, SA, P.Wsh0, 128, 8, P.bsh0, 128, nullptr, pb);
  // shading[1]: A cols 0..127 -> sin -> B cols 0..127
  do_layer<4, EPI_SIN>(bufA, SA, 0, bufB, SB, P.Wsh1, 128, 8, P.bsh1, 128, nullptr, pb);
  // shading[2]: B cols 0..127 -> sigmoid
  do_layer<4, EPI_SIG1>(bufB, SB, 0, nullptr, 0, P.Wsh2, 16, 1, P.bsh2, 1, o_shad, pb);
}

// ---------------------------------------------------------------- launch
extern "C" void kernel_launch(void* const* d_in, const int* in_sizes, int n_in,
                              void* d_out, int out_size, void* d_ws, size_t ws_size,
                              hipStream_t stream) {
  (void)n_in; (void)out_size; (void)ws_size;
  // setup_inputs order: x, w, then params leaves (alphabetical dict keys,
  // each (W, b) tuple flattened W-first):
  const float* x      = (const float*)d_in[0];
  const float* w      = (const float*)d_in[1];
  const float* W_alb  = (const float*)d_in[2];
  const float* b_alb  = (const float*)d_in[3];
  const float* W_amb0 = (const float*)d_in[4];
  const float* b_amb0 = (const float*)d_in[5];
  const float* W_amb1 = (const float*)d_in[6];
  const float* b_amb1 = (const float*)d_in[7];
  const float* W_mid  = (const float*)d_in[8];
  const float* b_mid  = (const float*)d_in[9];
  const float* W_sh0  = (const float*)d_in[10];
  const float* b_sh0  = (const float*)d_in[11];
  const float* W_sh1  = (const float*)d_in[12];
  const float* b_sh1  = (const float*)d_in[13];
  const float* W_sh2  = (const float*)d_in[14];
  const float* b_sh2  = (const float*)d_in[15];
  const float* W_sp[8];
  const float* b_sp[8];
  for (int i = 0; i < 8; ++i) {
    W_sp[i] = (const float*)d_in[16 + 2 * i];
    b_sp[i] = (const float*)d_in[17 + 2 * i];
  }
  const float* W_unc = (const float*)d_in[32];
  const float* b_unc = (const float*)d_in[33];

  int n = in_sizes[0] / 3;

  // workspace layout: packed f16 weights
  f16* wp = (f16*)d_ws;
  size_t off = 0;
  auto alloc = [&](size_t halves) { f16* p = wp + off; off += halves; return p; };
  f16* pWsp[8];
  pWsp[0] = alloc(64 * 256);
  for (int i = 1; i < 8; ++i) pWsp[i] = alloc(256 * 256);
  f16* pWmid  = alloc(256 * 256);
  f16* pWalb  = alloc(256 * 16);
  f16* pWunc  = alloc(64 * 16);
  f16* pWamb0 = alloc(64 * 128);
  f16* pWamb1 = alloc(128 * 16);
  f16* pWsh0  = alloc(320 * 128);
  f16* pWsh1  = alloc(128 * 128);
  f16* pWsh2  = alloc(128 * 16);

  auto pack = [&](const float* src, f16* dst, int K, int N, int Kp, int Np) {
    int total = Kp * Np;
    pack_w<<<(total + 255) / 256, 256, 0, stream>>>(src, dst, K, N, Kp, Np);
  };
  pack(W_sp[0], pWsp[0], 60, 256, 64, 256);
  for (int i = 1; i < 8; ++i) pack(W_sp[i], pWsp[i], 256, 256, 256, 256);
  pack(W_mid,  pWmid,  255, 256, 256, 256);
  pack(W_alb,  pWalb,  256, 3,   256, 16);
  pack(W_unc,  pWunc,  60,  1,   64,  16);
  pack(W_amb0, pWamb0, 60,  128, 64,  128);
  pack(W_amb1, pWamb1, 128, 3,   128, 16);
  pack(W_sh0,  pWsh0,  316, 128, 320, 128);
  pack(W_sh1,  pWsh1,  128, 128, 128, 128);
  pack(W_sh2,  pWsh2,  128, 1,   128, 16);

  NerfParams P;
  P.x = x; P.w = w;
  for (int i = 0; i < 8; ++i) { P.Wsp[i] = pWsp[i]; P.bsp[i] = b_sp[i]; }
  P.Wmid = pWmid;   P.bmid = b_mid;
  P.Walb = pWalb;   P.balb = b_alb;
  P.Wunc = pWunc;   P.bunc = b_unc;
  P.Wamb0 = pWamb0; P.bamb0 = b_amb0;
  P.Wamb1 = pWamb1; P.bamb1 = b_amb1;
  P.Wsh0 = pWsh0;   P.bsh0 = b_sh0;
  P.Wsh1 = pWsh1;   P.bsh1 = b_sh1;
  P.Wsh2 = pWsh2;   P.bsh2 = b_sh2;
  P.out = (float*)d_out;
  P.n = n;

  (void)hipFuncSetAttribute(reinterpret_cast<const void*>(nerf_main),
                            hipFuncAttributeMaxDynamicSharedMemorySize, LDS_BYTES);
  int blocks = n / (ROWS * WAVES);   // 524288 / 256 = 2048
  nerf_main<<<blocks, WAVES * 32, LDS_BYTES, stream>>>(P);
}
